// Attention_63170378989901
// MI455X (gfx1250) — matrix operand order repaired and checked by reading, not compiled, not run on hardware
//
#include <hip/hip_runtime.h>
#include <hip/hip_bf16.h>

#define NUM_HEADS   32
#define NUM_KV      8
#define HEAD_DIM    128
#define SEQ_T       2048
#define SCALING     0.08838834764831845f

#define BLOCK_M     128              // Q rows per block (8 waves x 16)
#define NWAVES      8

typedef __attribute__((ext_vector_type(16))) __bf16 v16bf;
typedef __attribute__((ext_vector_type(8)))  __bf16 v8bf;
typedef __attribute__((ext_vector_type(8)))  float  v8f;

// Pack two floats into two bf16 in one dword (fuses to v_cvt_pk_bf16_f32).
__device__ __forceinline__ unsigned pack_bf16(float a, float b) {
    unsigned short ua = __builtin_bit_cast(unsigned short, (__bf16)a);
    unsigned short ub = __builtin_bit_cast(unsigned short, (__bf16)b);
    return (unsigned)ua | ((unsigned)ub << 16);
}

__device__ __forceinline__ v16bf cat8(v8bf lo, v8bf hi) {
    return __builtin_shufflevector(lo, hi, 0, 1, 2, 3, 4, 5, 6, 7,
                                          8, 9, 10, 11, 12, 13, 14, 15);
}

// Max-reduce + broadcast across each 16-lane half using pure-VALU DPP:
// xor1 = quad_perm(1,0,3,2), xor2 = quad_perm(2,3,0,1),
// xor4 = DPP8 sel[i]=i^4, xor8 = row_ror:8 (== xor8 within a 16-lane row).
__device__ __forceinline__ float fmax_bcast16(float t) {
    int i;
    i = __builtin_amdgcn_update_dpp(0, __builtin_bit_cast(int, t), 0xB1, 0xF, 0xF, true);
    t = fmaxf(t, __builtin_bit_cast(float, i));
    i = __builtin_amdgcn_update_dpp(0, __builtin_bit_cast(int, t), 0x4E, 0xF, 0xF, true);
    t = fmaxf(t, __builtin_bit_cast(float, i));
    i = __builtin_amdgcn_mov_dpp8(__builtin_bit_cast(int, t), 6852524); // sel=i^4
    t = fmaxf(t, __builtin_bit_cast(float, i));
    i = __builtin_amdgcn_update_dpp(0, __builtin_bit_cast(int, t), 0x128, 0xF, 0xF, true);
    t = fmaxf(t, __builtin_bit_cast(float, i));
    return t;
}

// Flash attention, causal, GQA. Block = 256 threads (8 waves, wave32).
// Each wave handles a 16-row Q stripe; each 32-row KV tile staged in LDS as
// bf16 (V transposed) serves all 128 Q rows of the block. Softmax denominator
// via ones-column WMMA; DPP max; fragment ds_loads pipelined one step ahead.
__global__ __launch_bounds__(256) void fa_fwd_kernel(
    const float* __restrict__ q,
    const float* __restrict__ k,
    const float* __restrict__ v,
    float* __restrict__ out)
{
    __shared__ __bf16 Kb[32][136];        // K tile, row-major [kvrow][dim]
    __shared__ __bf16 Vt[128][40];        // V tile, TRANSPOSED [dim][kvrow]
    __shared__ __bf16 Pb[NWAVES][16][40]; // per-wave P (16 x 32) bf16

    const int wave   = threadIdx.x >> 5;
    const int lane   = threadIdx.x & 31;
    const int lane16 = lane & 15;
    const int hi     = lane >> 4;

    const int qblock = blockIdx.x;
    const int h      = blockIdx.y;
    const int b      = blockIdx.z;
    const int kvh    = h >> 2;               // GQA: H/KV = 4

    const int qrow0  = qblock * BLOCK_M + wave * 16;

    // ---- Q fragments, A layout, SCALING folded into the bf16 convert ----
    v16bf qa[4];
    {
        const float* qrow =
            q + (((size_t)b * SEQ_T + (qrow0 + lane16)) * NUM_HEADS + h) * HEAD_DIM;
        #pragma unroll
        for (int kk = 0; kk < 4; ++kk) {
            const int base = kk * 32 + (hi ? 16 : 0);
            #pragma unroll
            for (int e = 0; e < 16; ++e) {
                const int dim = base + e;
                qa[kk][e] = (__bf16)(qrow[dim] * SCALING);
            }
        }
    }

    // all-ones B fragment for the row-sum WMMA
    v16bf vones;
    #pragma unroll
    for (int e = 0; e < 16; ++e) vones[e] = (__bf16)1.0f;

    // ---- Accumulators ----
    v8f o[8] = {};
    v8f lsum = {};                // row-sum accumulator (same value in all cols)
    float mrow[8];
    #pragma unroll
    for (int r = 0; r < 8; ++r) mrow[r] = -1e30f;

    const int    ntiles  = qblock * 4 + 4;   // cover kv rows [0, (qblock+1)*128)
    const size_t rstride = (size_t)NUM_KV * HEAD_DIM;
    const size_t kvbase0 = ((size_t)b * SEQ_T * NUM_KV + kvh) * HEAD_DIM;

    for (int tile = 0; tile < ntiles; ++tile) {
        const int kv0 = tile * 32;

        __syncthreads();                 // previous-iteration LDS reads done

        // ---- Stage K (bf16 row-major) and V (bf16 transposed) ----
        {
            const float* kb = k + kvbase0 + (size_t)kv0 * rstride;
            const float* vb = v + kvbase0 + (size_t)kv0 * rstride;
            #pragma unroll
            for (int j = 0; j < 4; ++j) {        // K: 32 rows x 32 float4
                const int i = threadIdx.x + j * 256;
                const int row = i >> 5, c4 = (i & 31) * 4;
                const float4 kq = *(const float4*)(kb + (size_t)row * rstride + c4);
                uint2 pk;
                pk.x = pack_bf16(kq.x, kq.y);
                pk.y = pack_bf16(kq.z, kq.w);
                *(uint2*)&Kb[row][c4] = pk;
            }
            #pragma unroll
            for (int j = 0; j < 2; ++j) {        // V: 16 row-pairs x 32 float4
                const int i = threadIdx.x + j * 256;
                const int rp = i >> 5, c4 = (i & 31) * 4, row0 = rp * 2;
                const float4 a0 = *(const float4*)(vb + (size_t)(row0)     * rstride + c4);
                const float4 a1 = *(const float4*)(vb + (size_t)(row0 + 1) * rstride + c4);
                *(unsigned*)&Vt[c4 + 0][row0] = pack_bf16(a0.x, a1.x);
                *(unsigned*)&Vt[c4 + 1][row0] = pack_bf16(a0.y, a1.y);
                *(unsigned*)&Vt[c4 + 2][row0] = pack_bf16(a0.z, a1.z);
                *(unsigned*)&Vt[c4 + 3][row0] = pack_bf16(a0.w, a1.w);
            }
        }
        __syncthreads();

        if (kv0 > qrow0 + 15) continue;  // fully-masked for this wave (uniform)

        // ---- S = Q @ K^T, fragment loads pipelined one step ahead ----
        v8f s0 = {}, s1 = {};
        v16bf b0c, b1c;
        {
            const int dbase = hi * 16;
            b0c = cat8(*(const v8bf*)&Kb[lane16][dbase],
                       *(const v8bf*)&Kb[lane16][dbase + 8]);
            b1c = cat8(*(const v8bf*)&Kb[16 + lane16][dbase],
                       *(const v8bf*)&Kb[16 + lane16][dbase + 8]);
        }
        #pragma unroll
        for (int kk = 0; kk < 4; ++kk) {
            v16bf b0n, b1n;
            if (kk < 3) {
                const int dbase = (kk + 1) * 32 + hi * 16;
                b0n = cat8(*(const v8bf*)&Kb[lane16][dbase],
                           *(const v8bf*)&Kb[lane16][dbase + 8]);
                b1n = cat8(*(const v8bf*)&Kb[16 + lane16][dbase],
                           *(const v8bf*)&Kb[16 + lane16][dbase + 8]);
            }
            s0 = __builtin_amdgcn_wmma_f32_16x16x32_bf16(
                     false, qa[kk], false, b0c, (short)0, s0, false, false);
            s1 = __builtin_amdgcn_wmma_f32_16x16x32_bf16(
                     false, qa[kk], false, b1c, (short)0, s1, false, false);
            b0c = b0n; b1c = b1n;
        }

        // ---- Online softmax: DPP max, P -> LDS bf16, alpha per row ----
        float alpha[8];
        const int col0 = kv0 + lane16;
        const int col1 = col0 + 16;
        auto softmax_pass = [&](bool domask) {
            #pragma unroll
            for (int r = 0; r < 8; ++r) {
                float a  = s0[r];
                float bb = s1[r];
                if (domask) {
                    const int row = qrow0 + hi * 8 + r;
                    if (col0 > row) a  = -1e30f;
                    if (col1 > row) bb = -1e30f;
                }
                const float t    = fmax_bcast16(fmaxf(a, bb));
                const float mnew = fmaxf(mrow[r], t);
                alpha[r] = __expf(mrow[r] - mnew);
                mrow[r]  = mnew;
                Pb[wave][hi * 8 + r][lane16]      = (__bf16)__expf(a  - mnew);
                Pb[wave][hi * 8 + r][16 + lane16] = (__bf16)__expf(bb - mnew);
            }
        };
        if (kv0 + 31 > qrow0) softmax_pass(true);
        else                  softmax_pass(false);

        // ---- Rescale accumulators by alpha ----
        #pragma unroll
        for (int n = 0; n < 8; ++n)
            #pragma unroll
            for (int r = 0; r < 8; ++r)
                o[n][r] *= alpha[r];
        #pragma unroll
        for (int r = 0; r < 8; ++r)
            lsum[r] *= alpha[r];

        // ---- P back in A layout; lsum += P@1; O += P@V (pipelined) ----
        const v16bf pa = cat8(*(const v8bf*)&Pb[wave][lane16][hi * 16],
                              *(const v8bf*)&Pb[wave][lane16][hi * 16 + 8]);

        v16bf vfc = cat8(*(const v8bf*)&Vt[lane16][hi * 16],
                         *(const v8bf*)&Vt[lane16][hi * 16 + 8]);
        lsum = __builtin_amdgcn_wmma_f32_16x16x32_bf16(
                   false, pa, false, vones, (short)0, lsum, false, false);
        #pragma unroll
        for (int n = 0; n < 8; ++n) {
            v16bf vfn;
            if (n < 7) {
                const int col = (n + 1) * 16 + lane16;
                vfn = cat8(*(const v8bf*)&Vt[col][hi * 16],
                           *(const v8bf*)&Vt[col][hi * 16 + 8]);
            }
            o[n] = __builtin_amdgcn_wmma_f32_16x16x32_bf16(
                       false, pa, false, vfc, (short)0, o[n], false, false);
            vfc = vfn;
        }
    }

    // ---- Epilogue: normalize (lsum holds l in every lane) and store ----
    #pragma unroll
    for (int r = 0; r < 8; ++r) {
        const float inv = 1.0f / lsum[r];
        #pragma unroll
        for (int n = 0; n < 8; ++n)
            o[n][r] *= inv;
    }
    #pragma unroll
    for (int n = 0; n < 8; ++n) {
        const int col = n * 16 + lane16;
        #pragma unroll
        for (int r = 0; r < 8; ++r) {
            const int t = qrow0 + hi * 8 + r;
            out[(((size_t)b * SEQ_T + t) * NUM_HEADS + h) * HEAD_DIM + col] = o[n][r];
        }
    }
}

extern "C" void kernel_launch(void* const* d_in, const int* in_sizes, int n_in,
                              void* d_out, int out_size, void* d_ws, size_t ws_size,
                              hipStream_t stream) {
    (void)in_sizes; (void)n_in; (void)out_size; (void)d_ws; (void)ws_size;
    const float* q = (const float*)d_in[0];
    const float* k = (const float*)d_in[1];
    const float* v = (const float*)d_in[2];
    float* out = (float*)d_out;

    dim3 grid(SEQ_T / BLOCK_M, NUM_HEADS, 2 /*B*/);
    dim3 block(NWAVES * 32);
    fa_fwd_kernel<<<grid, block, 0, stream>>>(q, k, v, out);
}
